// MyDecoderLayerLKAFreq_21079699489295
// MI455X (gfx1250) — matrix operand order repaired and checked by reading, not compiled, not run on hardware
//
#include <hip/hip_runtime.h>

// ---------------------------------------------------------------------------
// Windowed cross-attention (LKA-Freq decoder layer) for gfx1250 / MI455X.
// One workgroup (256 threads = 8 wave32) per 8x8 window. All GEMMs use
// v_wmma_f32_16x16x32_f16. Weights pre-transposed fp32->fp16 [n][k] in d_ws
// so every B-fragment is two contiguous b128 loads per lane.
// ---------------------------------------------------------------------------

#define H_IMG   256
#define W_IMG   256
#define CDIM    256
#define WINSZ   8
#define NTOK    64          // 8*8 tokens per window
#define HEADS   8
#define HDIM    32
#define SCALE_Q 0.17677669529663687f   // 32^-0.5

typedef __attribute__((ext_vector_type(16))) _Float16 v16h;
typedef __attribute__((ext_vector_type(8)))  float    v8f;

__device__ __forceinline__ v8f wmma16(v16h a, v16h b, v8f c) {
  // D = A(16x32 f16) * B(32x16 f16) + C(16x16 f32)
  return __builtin_amdgcn_wmma_f32_16x16x32_f16(false, a, false, b,
                                                (short)0, c, false, false);
}

// ---- Fragment loaders (contiguous k = 16h + s order for A and B, wave32) --------------------
// A 16x32 (MxK): lane m = L%16. lanes<16: elems 0..15 = K0..15.
//                lanes>=16: elems 0..15 = K16..31.
__device__ __forceinline__ v16h frag_a(const _Float16* s, int row0, int col0,
                                       int ld, int lane) {
  const _Float16* p = s + (row0 + (lane & 15)) * ld + col0 + ((lane & 16) ? 16 : 0);
  union { v16h h; int4 q[2]; } u;
  u.q[0] = *(const int4*)(p);       // K elems 0..7
  u.q[1] = *(const int4*)(p + 8);   // K elems 8..15
  return u.h;
}

// B 32x16 (KxN) from an N-major source: B[k][n] = s[(row0+n)*ld + col0+k].
// lanes<16 hold K=0..15, lanes>=16 hold K=16..31 -> 16 contiguous halves.
__device__ __forceinline__ v16h frag_bt(const _Float16* s, int row0, int col0,
                                        int ld, int lane) {
  const _Float16* p = s + (row0 + (lane & 15)) * ld + col0 + ((lane & 16) ? 16 : 0);
  union { v16h h; int4 q[2]; } u;
  u.q[0] = *(const int4*)(p);
  u.q[1] = *(const int4*)(p + 8);
  return u.h;
}

// C 16x16 f32 store to f16 LDS (row-major dest): VGPR j -> row j / j+8
__device__ __forceinline__ void store_c(v8f acc, _Float16* dst, int ld,
                                        int row0, int col, float bias,
                                        float scale, int lane) {
  const float* f = (const float*)&acc;
  const int r = row0 + ((lane >> 4) << 3);
#pragma unroll
  for (int j = 0; j < 8; ++j)
    dst[(r + j) * ld + col] = (_Float16)((f[j] + bias) * scale);
}

// C store into transposed [ch][token] dest: 8 contiguous halves per lane
__device__ __forceinline__ void store_vt(v8f acc, _Float16* dst, int ch,
                                         int mt, float bias, int lane) {
  const float* f = (const float*)&acc;
  _Float16* p = dst + ch * NTOK + mt * 16 + ((lane >> 4) << 3);
#pragma unroll
  for (int j = 0; j < 8; ++j) p[j] = (_Float16)(f[j] + bias);
}

// C store to global NCHW output (window reverse fused): per lane one channel,
// rows rbase..rbase+7 -> contiguous 8-float runs in x.
__device__ __forceinline__ void store_out(v8f acc, float* op, int mt,
                                          float bias, int lane) {
  const float* f = (const float*)&acc;
  const int rbase = (lane >> 4) << 3;
#pragma unroll
  for (int j = 0; j < 8; ++j) {
    const int token = mt * 16 + rbase + j;
    op[(token >> 3) * W_IMG + (token & 7)] = f[j] + bias;
  }
}

// ---------------------------------------------------------------------------
// fp32 [k][n] weights -> fp16 transposed [n][k] in workspace
__global__ void convert_weights_kernel(const float* __restrict__ wq,
                                       const float* __restrict__ wkv,
                                       const float* __restrict__ wproj,
                                       _Float16* __restrict__ wq_t,
                                       _Float16* __restrict__ wkv_t,
                                       _Float16* __restrict__ wproj_t) {
  const int i = blockIdx.x * blockDim.x + threadIdx.x;   // 0..131071
  if (i < 65536) {
    const int k = i >> 8, n = i & 255;                   // [256][256]
    wq_t[n * 256 + k]    = (_Float16)wq[i];
    wproj_t[n * 256 + k] = (_Float16)wproj[i];
  }
  {
    const int k = i >> 9, n = i & 511;                   // [256][512]
    wkv_t[n * 256 + k] = (_Float16)wkv[i];
  }
}

// ---------------------------------------------------------------------------
__global__ __launch_bounds__(256) void win_attn_kernel(
    const float* __restrict__ qf,     // [B,256,256,256] NCHW
    const float* __restrict__ kvf,    // [B,256,256,256] NCHW
    const float* __restrict__ wdw,    // [256,1,3,3]
    const float* __restrict__ bq,     // [256]
    const float* __restrict__ bkv,    // [512]
    const float* __restrict__ bproj,  // [256]
    const _Float16* __restrict__ wq_t,    // [n=256][k=256]
    const _Float16* __restrict__ wkv_t,   // [n=512][k=256]
    const _Float16* __restrict__ wproj_t, // [n=256][k=256]
    float* __restrict__ out)          // [B,256,256,256] NCHW
{
  // 4 x (64x256) f16 buffers + 8 x (16x64) f16 softmax scratch = 144 KB
  __shared__ __align__(16) _Float16 smem[4 * NTOK * CDIM + HEADS * 16 * NTOK];
  _Float16* bufXqK  = smem;              // Xq, later K   [token][ch]
  _Float16* bufXkvO = smem + 16384;      // Xkv(conv), later O  [token][ch]
  _Float16* bufQ    = smem + 32768;      // Q (scaled)    [token][ch]
  _Float16* bufV    = smem + 49152;      // V transposed  [ch][token]
  _Float16* bufP    = smem + 65536;      // softmax probs [8][16][64]

  const int tid  = threadIdx.x;
  const int wave = tid >> 5;
  const int lane = tid & 31;
  const int wid  = blockIdx.x;           // 0..2047
  const int bN   = wid >> 10;            // batch
  const int rem  = wid & 1023;
  const int y0   = (rem >> 5) * WINSZ;
  const int x0   = (rem & 31) * WINSZ;

  // ======================= Stage 1: load window, dw-conv ====================
  {
    const int c = tid;                   // one channel per thread
    const float* qp = qf + (((size_t)(bN * CDIM + c)) * H_IMG + y0) * W_IMG + x0;
#pragma unroll
    for (int ty = 0; ty < 8; ++ty) {
      const float4 v0 = *(const float4*)(qp + ty * W_IMG);
      const float4 v1 = *(const float4*)(qp + ty * W_IMG + 4);
      _Float16* d = bufXqK + (ty * 8) * CDIM + c;
      d[0]        = (_Float16)v0.x; d[CDIM]     = (_Float16)v0.y;
      d[2 * CDIM] = (_Float16)v0.z; d[3 * CDIM] = (_Float16)v0.w;
      d[4 * CDIM] = (_Float16)v1.x; d[5 * CDIM] = (_Float16)v1.y;
      d[6 * CDIM] = (_Float16)v1.z; d[7 * CDIM] = (_Float16)v1.w;
    }
    float w9[9];
#pragma unroll
    for (int i = 0; i < 9; ++i) w9[i] = wdw[c * 9 + i];
    const float* kp = kvf + ((size_t)(bN * CDIM + c)) * (H_IMG * W_IMG);
    float r0[10], r1[10], r2[10];
    auto load_row = [&](float* r, int y) {
      if ((unsigned)y < (unsigned)H_IMG) {
        const float* p = kp + y * W_IMG + x0;
        r[0] = (x0 > 0) ? p[-1] : 0.f;
        const float4 a  = *(const float4*)p;
        const float4 b2 = *(const float4*)(p + 4);
        r[1] = a.x;  r[2] = a.y;  r[3] = a.z;  r[4] = a.w;
        r[5] = b2.x; r[6] = b2.y; r[7] = b2.z; r[8] = b2.w;
        r[9] = (x0 + 8 < W_IMG) ? p[8] : 0.f;
      } else {
#pragma unroll
        for (int i = 0; i < 10; ++i) r[i] = 0.f;
      }
    };
    load_row(r0, y0 - 1);
    load_row(r1, y0);
#pragma unroll
    for (int ty = 0; ty < 8; ++ty) {
      load_row(r2, y0 + ty + 1);
#pragma unroll
      for (int tx = 0; tx < 8; ++tx) {
        const float acc =
            w9[0] * r0[tx] + w9[1] * r0[tx + 1] + w9[2] * r0[tx + 2] +
            w9[3] * r1[tx] + w9[4] * r1[tx + 1] + w9[5] * r1[tx + 2] +
            w9[6] * r2[tx] + w9[7] * r2[tx + 1] + w9[8] * r2[tx + 2];
        bufXkvO[(ty * 8 + tx) * CDIM + c] = (_Float16)acc;
      }
#pragma unroll
      for (int i = 0; i < 10; ++i) { r0[i] = r1[i]; r1[i] = r2[i]; }
    }
  }
  __syncthreads();

  // ======================= Stage 2a: Q = Xq @ Wq  (2 cols/wave jointly) =====
  {
    const int nA = wave * 16, nB = (wave + 8) * 16;
    v8f qa0 = {}, qa1 = {}, qa2 = {}, qa3 = {};
    v8f qb0 = {}, qb1 = {}, qb2 = {}, qb3 = {};
#pragma unroll
    for (int ks = 0; ks < 8; ++ks) {
      const int k0 = ks * 32;
      const v16h a0 = frag_a(bufXqK, 0,  k0, 256, lane);
      const v16h a1 = frag_a(bufXqK, 16, k0, 256, lane);
      const v16h a2 = frag_a(bufXqK, 32, k0, 256, lane);
      const v16h a3 = frag_a(bufXqK, 48, k0, 256, lane);
      const v16h bA = frag_bt(wq_t, nA, k0, 256, lane);
      const v16h bB = frag_bt(wq_t, nB, k0, 256, lane);
      qa0 = wmma16(a0, bA, qa0); qa1 = wmma16(a1, bA, qa1);
      qa2 = wmma16(a2, bA, qa2); qa3 = wmma16(a3, bA, qa3);
      qb0 = wmma16(a0, bB, qb0); qb1 = wmma16(a1, bB, qb1);
      qb2 = wmma16(a2, bB, qb2); qb3 = wmma16(a3, bB, qb3);
    }
    const int   cA = nA + (lane & 15), cB = nB + (lane & 15);
    const float bA = bq[cA], bB = bq[cB];
    store_c(qa0, bufQ, 256, 0,  cA, bA, SCALE_Q, lane);
    store_c(qa1, bufQ, 256, 16, cA, bA, SCALE_Q, lane);
    store_c(qa2, bufQ, 256, 32, cA, bA, SCALE_Q, lane);
    store_c(qa3, bufQ, 256, 48, cA, bA, SCALE_Q, lane);
    store_c(qb0, bufQ, 256, 0,  cB, bB, SCALE_Q, lane);
    store_c(qb1, bufQ, 256, 16, cB, bB, SCALE_Q, lane);
    store_c(qb2, bufQ, 256, 32, cB, bB, SCALE_Q, lane);
    store_c(qb3, bufQ, 256, 48, cB, bB, SCALE_Q, lane);
  }
  __syncthreads();

  // ======================= Stage 2b: [K|V] = Xkv @ Wkv ======================
  // pair 0 -> K columns (0..255, row-major into bufXqK)
  // pair 1 -> V columns (256..511, transposed into bufV)
#pragma unroll
  for (int p = 0; p < 2; ++p) {
    const int nA = (wave + 16 * p) * 16;       // column in 512-wide kv output
    const int nB = (wave + 8 + 16 * p) * 16;
    v8f va0 = {}, va1 = {}, va2 = {}, va3 = {};
    v8f vb0 = {}, vb1 = {}, vb2 = {}, vb3 = {};
#pragma unroll
    for (int ks = 0; ks < 8; ++ks) {
      const int k0 = ks * 32;
      const v16h a0 = frag_a(bufXkvO, 0,  k0, 256, lane);
      const v16h a1 = frag_a(bufXkvO, 16, k0, 256, lane);
      const v16h a2 = frag_a(bufXkvO, 32, k0, 256, lane);
      const v16h a3 = frag_a(bufXkvO, 48, k0, 256, lane);
      const v16h bA = frag_bt(wkv_t, nA, k0, 256, lane);
      const v16h bB = frag_bt(wkv_t, nB, k0, 256, lane);
      va0 = wmma16(a0, bA, va0); va1 = wmma16(a1, bA, va1);
      va2 = wmma16(a2, bA, va2); va3 = wmma16(a3, bA, va3);
      vb0 = wmma16(a0, bB, vb0); vb1 = wmma16(a1, bB, vb1);
      vb2 = wmma16(a2, bB, vb2); vb3 = wmma16(a3, bB, vb3);
    }
    const float biasA = bkv[nA + (lane & 15)];
    const float biasB = bkv[nB + (lane & 15)];
    if (p == 0) {            // K, row-major [token][ch], overwrites Xq
      const int cA = nA + (lane & 15), cB = nB + (lane & 15);
      store_c(va0, bufXqK, 256, 0,  cA, biasA, 1.f, lane);
      store_c(va1, bufXqK, 256, 16, cA, biasA, 1.f, lane);
      store_c(va2, bufXqK, 256, 32, cA, biasA, 1.f, lane);
      store_c(va3, bufXqK, 256, 48, cA, biasA, 1.f, lane);
      store_c(vb0, bufXqK, 256, 0,  cB, biasB, 1.f, lane);
      store_c(vb1, bufXqK, 256, 16, cB, biasB, 1.f, lane);
      store_c(vb2, bufXqK, 256, 32, cB, biasB, 1.f, lane);
      store_c(vb3, bufXqK, 256, 48, cB, biasB, 1.f, lane);
    } else {                 // V, transposed [ch][token]
      const int chA = nA - 256 + (lane & 15);
      const int chB = nB - 256 + (lane & 15);
      store_vt(va0, bufV, chA, 0, biasA, lane);
      store_vt(va1, bufV, chA, 1, biasA, lane);
      store_vt(va2, bufV, chA, 2, biasA, lane);
      store_vt(va3, bufV, chA, 3, biasA, lane);
      store_vt(vb0, bufV, chB, 0, biasB, lane);
      store_vt(vb1, bufV, chB, 1, biasB, lane);
      store_vt(vb2, bufV, chB, 2, biasB, lane);
      store_vt(vb3, bufV, chB, 3, biasB, lane);
    }
  }
  __syncthreads();

  // ======================= Stage 3: attention (1 wave = 1 head) =============
  {
    const int  off = wave * HDIM;                  // head column offset
    _Float16*  P   = bufP + wave * (16 * 64);      // per-wave prob scratch
#pragma unroll 1
    for (int mb = 0; mb < 4; ++mb) {
      // S[16x64] = Qblk[16x32] @ K^T[32x64]   (K=32, one wmma per tile)
      const v16h aq = frag_a(bufQ, mb * 16, off, 256, lane);
      v8f s0 = {}, s1 = {}, s2 = {}, s3 = {};
      s0 = wmma16(aq, frag_bt(bufXqK, 0,  off, 256, lane), s0);
      s1 = wmma16(aq, frag_bt(bufXqK, 16, off, 256, lane), s1);
      s2 = wmma16(aq, frag_bt(bufXqK, 32, off, 256, lane), s2);
      s3 = wmma16(aq, frag_bt(bufXqK, 48, off, 256, lane), s3);
      const float* f0 = (const float*)&s0;
      const float* f1 = (const float*)&s1;
      const float* f2 = (const float*)&s2;
      const float* f3 = (const float*)&s3;
      const int rbase = (lane >> 4) << 3;
      // row softmax: row m lives across 16 lanes (one half-wave) x 4 tiles
#pragma unroll
      for (int j = 0; j < 8; ++j) {
        float mx = fmaxf(fmaxf(f0[j], f1[j]), fmaxf(f2[j], f3[j]));
#pragma unroll
        for (int m = 1; m < 16; m <<= 1) mx = fmaxf(mx, __shfl_xor(mx, m, 32));
        const float e0 = __expf(f0[j] - mx);
        const float e1 = __expf(f1[j] - mx);
        const float e2 = __expf(f2[j] - mx);
        const float e3 = __expf(f3[j] - mx);
        float sum = e0 + e1 + e2 + e3;
#pragma unroll
        for (int m = 1; m < 16; m <<= 1) sum += __shfl_xor(sum, m, 32);
        const float inv = 1.f / sum;
        _Float16* pr = P + (rbase + j) * 64 + (lane & 15);
        pr[0]  = (_Float16)(e0 * inv);
        pr[16] = (_Float16)(e1 * inv);
        pr[32] = (_Float16)(e2 * inv);
        pr[48] = (_Float16)(e3 * inv);
      }
      __builtin_amdgcn_wave_barrier();   // order P writes vs A-frag reads
      // O[16x32] = P[16x64] @ Vh[64x32]; V is [ch][token] -> frag_bt
#pragma unroll
      for (int nt = 0; nt < 2; ++nt) {
        v8f o = {};
        o = wmma16(frag_a(P, 0, 0,  64, lane),
                   frag_bt(bufV, off + nt * 16, 0,  64, lane), o);
        o = wmma16(frag_a(P, 0, 32, 64, lane),
                   frag_bt(bufV, off + nt * 16, 32, 64, lane), o);
        store_c(o, bufXkvO, 256, mb * 16, off + nt * 16 + (lane & 15),
                0.f, 1.f, lane);        // O overwrites Xkv
      }
    }
  }
  __syncthreads();

  // ======================= Stage 4: out = O @ Wproj + b, scatter NCHW ======
  {
    const int nA = wave * 16, nB = (wave + 8) * 16;
    v8f pa0 = {}, pa1 = {}, pa2 = {}, pa3 = {};
    v8f pb0 = {}, pb1 = {}, pb2 = {}, pb3 = {};
#pragma unroll
    for (int ks = 0; ks < 8; ++ks) {
      const int k0 = ks * 32;
      const v16h a0 = frag_a(bufXkvO, 0,  k0, 256, lane);
      const v16h a1 = frag_a(bufXkvO, 16, k0, 256, lane);
      const v16h a2 = frag_a(bufXkvO, 32, k0, 256, lane);
      const v16h a3 = frag_a(bufXkvO, 48, k0, 256, lane);
      const v16h bA = frag_bt(wproj_t, nA, k0, 256, lane);
      const v16h bB = frag_bt(wproj_t, nB, k0, 256, lane);
      pa0 = wmma16(a0, bA, pa0); pa1 = wmma16(a1, bA, pa1);
      pa2 = wmma16(a2, bA, pa2); pa3 = wmma16(a3, bA, pa3);
      pb0 = wmma16(a0, bB, pb0); pb1 = wmma16(a1, bB, pb1);
      pb2 = wmma16(a2, bB, pb2); pb3 = wmma16(a3, bB, pb3);
    }
    const int   chA = nA + (lane & 15), chB = nB + (lane & 15);
    const float bA = bproj[chA], bB = bproj[chB];
    float* opA = out + (((size_t)(bN * CDIM + chA)) * H_IMG + y0) * W_IMG + x0;
    float* opB = out + (((size_t)(bN * CDIM + chB)) * H_IMG + y0) * W_IMG + x0;
    store_out(pa0, opA, 0, bA, lane); store_out(pa1, opA, 1, bA, lane);
    store_out(pa2, opA, 2, bA, lane); store_out(pa3, opA, 3, bA, lane);
    store_out(pb0, opB, 0, bB, lane); store_out(pb1, opB, 1, bB, lane);
    store_out(pb2, opB, 2, bB, lane); store_out(pb3, opB, 3, bB, lane);
  }
}

// ---------------------------------------------------------------------------
extern "C" void kernel_launch(void* const* d_in, const int* in_sizes, int n_in,
                              void* d_out, int out_size, void* d_ws,
                              size_t ws_size, hipStream_t stream) {
  const float* qf    = (const float*)d_in[0];
  const float* kvf   = (const float*)d_in[1];
  const float* w_dw  = (const float*)d_in[2];
  const float* wq    = (const float*)d_in[3];
  const float* bq    = (const float*)d_in[4];
  const float* wkv   = (const float*)d_in[5];
  const float* bkv   = (const float*)d_in[6];
  const float* wproj = (const float*)d_in[7];
  const float* bproj = (const float*)d_in[8];
  float* out = (float*)d_out;

  _Float16* wq_t    = (_Float16*)d_ws;          // [256][256] halves
  _Float16* wkv_t   = wq_t + 65536;             // [512][256] halves
  _Float16* wproj_t = wkv_t + 131072;           // [256][256] halves

  convert_weights_kernel<<<512, 256, 0, stream>>>(wq, wkv, wproj,
                                                  wq_t, wkv_t, wproj_t);
  win_attn_kernel<<<2048, 256, 0, stream>>>(qf, kvf, w_dw, bq, bkv, bproj,
                                            wq_t, wkv_t, wproj_t, out);
}
